// TransformerBlock_3049426780431
// MI455X (gfx1250) — matrix operand order repaired and checked by reading, not compiled, not run on hardware
//
#include <hip/hip_runtime.h>

typedef __attribute__((ext_vector_type(16))) _Float16 v16h;
typedef __attribute__((ext_vector_type(8)))  _Float16 v8h;
typedef __attribute__((ext_vector_type(2)))  _Float16 v2h;
typedef __attribute__((ext_vector_type(8)))  float    v8f;

#define B_    8
#define N_    4096
#define KNN_  20
#define BN_   (B_ * N_)                 /* 32768 */
#define PTOT_ ((long long)BN_ * KNN_)   /* 655360 */
#define SGRID 1024
#define FS_   168   /* fcat LDS stride (halves): 84 dwords -> conflict-free b128 */
#define HS_   72    /* 64-chan LDS stride (halves): 36 dwords -> conflict-free b128 */

// ---------------- CDNA5 async copy (inline asm, ASYNCcnt-tracked) ----------------

__device__ inline void async_copy16(void* lds_dst, const void* gsrc) {
  unsigned l = (unsigned)(unsigned long long)lds_dst;
  unsigned long long g = (unsigned long long)gsrc;
  asm volatile("global_load_async_to_lds_b128 %0, %1, off" :: "v"(l), "v"(g) : "memory");
}
__device__ inline void async_copy32(void* lds_dst, const void* gsrc) {
  unsigned l = (unsigned)(unsigned long long)lds_dst;
  unsigned long long g = (unsigned long long)gsrc;
  asm volatile("global_load_async_to_lds_b128 %0, %1, off" :: "v"(l), "v"(g) : "memory");
  asm volatile("global_load_async_to_lds_b128 %0, %1, off offset:16" :: "v"(l), "v"(g) : "memory");
}
__device__ inline void wait_async() {
  asm volatile("s_wait_asynccnt 0x0" ::: "memory");
}

// ---------------- WMMA helpers (ISA 7.12.2 layouts) ----------------

__device__ inline v16h load_a16(const _Float16* __restrict__ W, int row_base, int ld, int kbase) {
  // 16-bit A matrix 16x32: lanes 0-15 row M=lane, K pairs {0,1}..{6,7},{16,17}..{22,23};
  // lanes 16-31: same rows, K offset +8. Two contiguous 16B runs -> global_load_b128 x2.
  int lane = threadIdx.x & 31;
  int row  = row_base + (lane & 15);
  int kh   = (lane >> 4) << 3;
  v16h a;
#pragma unroll
  for (int j = 0; j < 8; ++j) {
    int k0 = kbase + kh + ((j < 4) ? (2 * j) : (16 + 2 * (j - 4)));
    a[2 * j]     = W[row * ld + k0];
    a[2 * j + 1] = W[row * ld + k0 + 1];
  }
  return a;
}

__device__ inline v16h load_b16(const _Float16* __restrict__ S, int stride, int kbase) {
  // B matrix 32x16 from position-major LDS [p][stride]:
  // lanes 0-15: col = lane, K = kbase+{0..7,16..23} ; lanes 16-31: col-16, K += 8.
  // Two 16B runs per lane, 32B apart -> two ds_load_b128.
  int lane = threadIdx.x & 31;
  int col  = lane & 15;
  int kh   = (lane >> 4) << 3;
  const _Float16* r = S + col * stride + kbase + kh;
  v8h lo = *(const v8h*)(r);
  v8h hi = *(const v8h*)(r + 16);
  return __builtin_shufflevector(lo, hi, 0, 1, 2, 3, 4, 5, 6, 7, 8, 9, 10, 11, 12, 13, 14, 15);
}

__device__ inline void store_cd(float* __restrict__ dst, long long pos_base, int C,
                                int chan_base, v8f acc) {
  int lane = threadIdx.x & 31;
  int col  = lane & 15;
  int rb   = (lane >> 4) * 8;
#pragma unroll
  for (int r = 0; r < 8; ++r)
    dst[(pos_base + col) * C + chan_base + rb + r] = acc[r];
}

__device__ inline v8f wmma16(v16h a, v16h b, v8f c) {
  return __builtin_amdgcn_wmma_f32_16x16x32_f16(false, a, false, b, (short)0, c, false, false);
}

// ---------------- 1) weight conversion to padded f16 ----------------

__global__ void k_convw(const float* __restrict__ W1, const float* __restrict__ W2,
                        const float* __restrict__ Wv, const float* __restrict__ Wk,
                        const float* __restrict__ Wq,
                        _Float16* W1h, _Float16* W2h, _Float16* Wvh, _Float16* Wkh,
                        _Float16* Wqh) {
  int i = blockIdx.x * 256 + threadIdx.x;
  if (i < 64 * 160) { int r = i / 160, c = i % 160; W1h[i] = (_Float16)(c < 144 ? W1[r * 144 + c] : 0.f); }
  if (i < 64 * 64)  { W2h[i] = (_Float16)W2[i]; }
  if (i < 16 * 160) { int r = i / 160, c = i % 160; Wvh[i] = (_Float16)(c < 144 ? Wv[r * 144 + c] : 0.f); }
  if (i < 32 * 160) { int r = i / 160, c = i % 160; Wkh[i] = (_Float16)(c < 144 ? Wk[r * 144 + c] : 0.f); }
  if (i < 256 * 64) { Wqh[i] = (_Float16)Wq[i]; }
}

// ---------------- 2) pack xyz + sqnorm, emit xyz output ----------------

__global__ void k_pre(const float* __restrict__ x, float4* __restrict__ xyzw,
                      float* __restrict__ out_xyz) {
  int i = blockIdx.x * 256 + threadIdx.x;
  if (i < BN_) {
    const float* p = x + (long long)i * 67;
    float a = p[0], b = p[1], c = p[2];
    float4 f; f.x = a; f.y = b; f.z = c; f.w = a * a + b * b + c * c;
    xyzw[i] = f;
    out_xyz[3 * i] = a; out_xyz[3 * i + 1] = b; out_xyz[3 * i + 2] = c;
  }
}

// ---------------- 3) KNN: async-staged tiles + register top-20 ----------------

__global__ __launch_bounds__(128) void k_knn(const float4* __restrict__ xyzw,
                                             int* __restrict__ idx) {
  __shared__ float4 tile[256];
  int b   = blockIdx.x >> 5;                      // 32 blocks per batch
  int row = ((blockIdx.x & 31) * 128) + threadIdx.x;
  const float4* base = xyzw + b * N_;
  float4 me = base[row];
  float best[KNN_]; int bidx[KNN_];
#pragma unroll
  for (int j = 0; j < KNN_; ++j) { best[j] = 3.4e38f; bidx[j] = 0; }
  for (int t = 0; t < N_; t += 256) {
    // async bulk copy: 128 threads x 32B = 4KB candidate tile
    async_copy32(&tile[threadIdx.x * 2], base + t + threadIdx.x * 2);
    if (t + 256 < N_)
      __builtin_prefetch((const void*)(base + t + 256 + threadIdx.x * 2), 0, 0);
    wait_async();
    __syncthreads();
    for (int i = 0; i < 256; ++i) {
      float4 o = tile[i];
      float d = me.w + o.w - 2.f * (me.x * o.x + me.y * o.y + me.z * o.z);
      if (d < best[KNN_ - 1]) {
        float dd = d; int ii = t + i;
#pragma unroll
        for (int j = 0; j < KNN_; ++j) {
          if (dd < best[j]) {
            float tf = best[j]; best[j] = dd; dd = tf;
            int   ti = bidx[j]; bidx[j] = ii; ii = ti;
          }
        }
      }
    }
    __syncthreads();
  }
  long long o = ((long long)b * N_ + row) * KNN_;
#pragma unroll
  for (int j = 0; j < KNN_; ++j) idx[o + j] = bidx[j];
}

// ---------------- 4) fcat build (LDS, f16) + W1/Wk/Wv WMMA GEMMs ----------------

__global__ __launch_bounds__(128) void k_fcat_gemm(
    const float* __restrict__ x, const int* __restrict__ idx,
    const _Float16* __restrict__ W1h, const _Float16* __restrict__ Wkh,
    const _Float16* __restrict__ Wvh, const float* __restrict__ Wh,
    float* __restrict__ z1, float* __restrict__ zk, float* __restrict__ zv) {
  __shared__ _Float16 sf[16 * FS_];
  int tid = threadIdx.x;
  long long tile_base = (long long)blockIdx.x * 16;
  // 8 threads per position, paired-channel b32 stores, position-major layout
  {
    int p  = tid >> 3;
    int c0 = (tid & 7) * 2;
    long long pid = tile_base + p;
    int b  = (int)(pid / (N_ * KNN_));
    int r  = (int)(pid % (N_ * KNN_));
    int n  = r / KNN_;
    int nb = idx[pid];
    const float* xc = x + (long long)(b * N_ + n)  * 67;
    const float* xn = x + (long long)(b * N_ + nb) * 67;
    float rx = xn[0] - xc[0], ry = xn[1] - xc[1], rz = xn[2] - xc[2];
#pragma unroll
    for (int c = c0; c < 160; c += 16) {
      v2h hv;
#pragma unroll
      for (int u = 0; u < 2; ++u) {
        int cc = c + u;
        float val = 0.f;
        if (cc < 64)       val = xn[3 + cc] - xc[3 + cc];          // gpn
        else if (cc < 128) val = xc[3 + (cc - 64)];                // center feat
        else if (cc < 144) {                                       // pos = Wh @ rel
          int vv = cc - 128;
          val = Wh[vv * 3 + 0] * rx + Wh[vv * 3 + 1] * ry + Wh[vv * 3 + 2] * rz;
        }
        hv[u] = (_Float16)val;
      }
      *(v2h*)(sf + p * FS_ + c) = hv;
    }
  }
  __syncthreads();
  int w = tid >> 5;
  { // z1 : 64 out channels, wave w does channels [16w, 16w+16)
    v8f acc = {};
#pragma unroll
    for (int kc = 0; kc < 5; ++kc)
      acc = wmma16(load_a16(W1h, w * 16, 160, kc * 32), load_b16(sf, FS_, kc * 32), acc);
    store_cd(z1, tile_base, 64, w * 16, acc);
  }
  if (w < 2) { // zk : 32 out channels
    v8f acc = {};
#pragma unroll
    for (int kc = 0; kc < 5; ++kc)
      acc = wmma16(load_a16(Wkh, w * 16, 160, kc * 32), load_b16(sf, FS_, kc * 32), acc);
    store_cd(zk, tile_base, 32, w * 16, acc);
  } else if (w == 2) { // zv : 16 out channels
    v8f acc = {};
#pragma unroll
    for (int kc = 0; kc < 5; ++kc)
      acc = wmma16(load_a16(Wvh, 0, 160, kc * 32), load_b16(sf, FS_, kc * 32), acc);
    store_cd(zv, tile_base, 16, 0, acc);
  }
}

// ---------------- deterministic two-stage channel stats ----------------

template <int C>
__global__ void k_stats(const float* __restrict__ src, long long nelem,
                        float* __restrict__ psum, float* __restrict__ psq) {
  __shared__ float ls[256], lq[256];
  int tid = threadIdx.x;
  long long stride = (long long)gridDim.x * 256;
  float s = 0.f, q = 0.f;
  for (long long i = (long long)blockIdx.x * 256 + tid; i < nelem; i += stride) {
    float v = src[i];
    s += v; q += v * v;
  }
  ls[tid] = s; lq[tid] = q;
  __syncthreads();
  if (tid < C) { // thread j always sees channel j & (C-1): stride % C == 0
    float S = 0.f, Q = 0.f;
    for (int j = tid; j < 256; j += C) { S += ls[j]; Q += lq[j]; }
    psum[blockIdx.x * C + tid] = S;
    psq [blockIdx.x * C + tid] = Q;
  }
}

__global__ void k_finalize(const float* __restrict__ psum, const float* __restrict__ psq,
                           int C, int nblocks, float invM,
                           float* __restrict__ mu, float* __restrict__ rs) {
  int c = threadIdx.x;
  if (c < C) {
    float S = 0.f, Q = 0.f;
    for (int b = 0; b < nblocks; ++b) { S += psum[b * C + c]; Q += psq[b * C + c]; }
    float m = S * invM;
    float v = Q * invM - m * m;
    mu[c] = m;
    rs[c] = rsqrtf(v + 1e-5f);
  }
}

// ---------------- 6) BN+ReLU fused into W2 WMMA GEMM ----------------

__global__ __launch_bounds__(128) void k_bn1_gemm2(
    const float* __restrict__ z1, const float* __restrict__ mu1, const float* __restrict__ rs1,
    const float* __restrict__ g1, const float* __restrict__ b1,
    const _Float16* __restrict__ W2h, float* __restrict__ z2) {
  __shared__ _Float16 sh[16 * HS_];
  int tid = threadIdx.x;
  long long base = (long long)blockIdx.x * 16;
  {
    int p  = tid >> 3;
    int c0 = (tid & 7) * 2;
#pragma unroll
    for (int c = c0; c < 64; c += 16) {
      v2h hv;
#pragma unroll
      for (int u = 0; u < 2; ++u) {
        int cc = c + u;
        float v = z1[(base + p) * 64 + cc];
        v = (v - mu1[cc]) * rs1[cc] * g1[cc] + b1[cc];
        hv[u] = (_Float16)fmaxf(v, 0.f);
      }
      *(v2h*)(sh + p * HS_ + c) = hv;
    }
  }
  __syncthreads();
  int w = tid >> 5;
  v8f acc = {};
#pragma unroll
  for (int kc = 0; kc < 2; ++kc)
    acc = wmma16(load_a16(W2h, w * 16, 64, kc * 32), load_b16(sh, HS_, kc * 32), acc);
  store_cd(z2, base, 64, w * 16, acc);
}

// ---------------- 7) BN+ReLU + max over K -> fq (f16) ----------------

__global__ void k_fqmax(const float* __restrict__ z2, const float* __restrict__ mu2,
                        const float* __restrict__ rs2, const float* __restrict__ g2,
                        const float* __restrict__ b2, _Float16* __restrict__ fqh) {
  int bn = blockIdx.x;
  int c  = threadIdx.x; // 64
  float m = -1e30f;
  for (int k = 0; k < KNN_; ++k) {
    float v = z2[((long long)bn * KNN_ + k) * 64 + c];
    v = fmaxf((v - mu2[c]) * rs2[c] * g2[c] + b2[c], 0.f);
    m = fmaxf(m, v);
  }
  fqh[(long long)bn * 64 + c] = (_Float16)m;
}

// ---------------- 8) Wq (256x64) WMMA GEMM -> zq (async-staged LDS tile) ----------------

__global__ __launch_bounds__(128) void k_zq(const _Float16* __restrict__ fqh,
                                            const _Float16* __restrict__ Wqh,
                                            float* __restrict__ zq) {
  __shared__ _Float16 sh[16 * HS_];
  int tid = threadIdx.x;
  long long base = (long long)blockIdx.x * 16;
  {
    // bulk tile stage: each thread async-copies 8 halves (16B)
    int p = tid >> 3, ch = (tid & 7) * 8;
    async_copy16(sh + p * HS_ + ch, fqh + (base + p) * 64 + ch);
    wait_async();
  }
  __syncthreads();
  int w = tid >> 5;
#pragma unroll
  for (int mi = 0; mi < 4; ++mi) {
    int mt = w * 4 + mi;
    v8f acc = {};
#pragma unroll
    for (int kc = 0; kc < 2; ++kc)
      acc = wmma16(load_a16(Wqh, mt * 16, 64, kc * 32), load_b16(sh, HS_, kc * 32), acc);
    store_cd(zq, base, 256, mt * 16, acc);
  }
}

// ---------------- 9) final fusion: BN(q), softmax(k), BN(v), kv, q*kv ----------------

__global__ __launch_bounds__(128) void k_final(
    const float* __restrict__ zq, const float* __restrict__ muq, const float* __restrict__ rsq_,
    const float* __restrict__ gq, const float* __restrict__ bq,
    const float* __restrict__ zk, const float* __restrict__ zv,
    const float* __restrict__ muv, const float* __restrict__ rsv,
    const float* __restrict__ gv, const float* __restrict__ bv,
    float* __restrict__ out) {
  __shared__ float s_q[256];
  __shared__ float s_kk[KNN_ * 32];
  __shared__ float s_v[KNN_ * 16];
  __shared__ float s_kv[32 * 16];
  __shared__ float s_m[32], s_s[32];
  int bn  = blockIdx.x;
  int tid = threadIdx.x; // 128
  for (int c = tid; c < 256; c += 128) {
    float v = (zq[(long long)bn * 256 + c] - muq[c]) * rsq_[c] * gq[c] + bq[c];
    s_q[c] = fmaxf(v, 0.f);
  }
  for (int e = tid; e < KNN_ * 32; e += 128)
    s_kk[e] = zk[(long long)bn * (KNN_ * 32) + e];
  for (int e = tid; e < KNN_ * 16; e += 128) {
    int vv = e & 15;
    float v = zv[(long long)bn * (KNN_ * 16) + e];
    s_v[e] = (v - muv[vv]) * rsv[vv] * gv[vv] + bv[vv];
  }
  __syncthreads();
  if (tid < 32) {
    float m = -1e30f;
    for (int k = 0; k < KNN_; ++k) m = fmaxf(m, s_kk[k * 32 + tid]);
    float s = 0.f;
    for (int k = 0; k < KNN_; ++k) s += __expf(s_kk[k * 32 + tid] - m);
    s_m[tid] = m; s_s[tid] = 1.f / s;
  }
  __syncthreads();
  for (int e = tid; e < KNN_ * 32; e += 128) {
    int d = e & 31;
    s_kk[e] = __expf(s_kk[e] - s_m[d]) * s_s[d];
  }
  __syncthreads();
  for (int e = tid; e < 32 * 16; e += 128) {
    int d = e >> 4, vv = e & 15;
    float s = 0.f;
#pragma unroll
    for (int k = 0; k < KNN_; ++k) s += s_kk[k * 32 + d] * s_v[k * 16 + vv];
    s_kv[e] = s;
  }
  __syncthreads();
  {
    int h = tid >> 4, vv = tid & 15;
    float s = 0.f;
#pragma unroll
    for (int d = 0; d < 32; ++d) s += s_q[h * 32 + d] * s_kv[d * 16 + vv];
    int b = bn >> 12, n = bn & (N_ - 1);
    out[((long long)b * 128 + tid) * N_ + n] = s;
  }
}

// ---------------- host launcher ----------------

extern "C" void kernel_launch(void* const* d_in, const int* in_sizes, int n_in,
                              void* d_out, int out_size, void* d_ws, size_t ws_size,
                              hipStream_t stream) {
  (void)in_sizes; (void)n_in; (void)out_size; (void)ws_size;
  const float* x  = (const float*)d_in[0];
  const float* W1 = (const float*)d_in[1];
  const float* g1 = (const float*)d_in[2];
  const float* b1 = (const float*)d_in[3];
  const float* W2 = (const float*)d_in[4];
  const float* g2 = (const float*)d_in[5];
  const float* b2 = (const float*)d_in[6];
  const float* Wv = (const float*)d_in[7];
  const float* gv = (const float*)d_in[8];
  const float* bv = (const float*)d_in[9];
  const float* Wk = (const float*)d_in[10];
  const float* Wq = (const float*)d_in[11];
  const float* gq = (const float*)d_in[12];
  const float* bq = (const float*)d_in[13];
  const float* Wh = (const float*)d_in[14];

  char* ws = (char*)d_ws;
  size_t off = 0;
  auto alloc = [&](size_t bytes) -> size_t {
    size_t r = off;
    off += (bytes + 255) & ~(size_t)255;
    return r;
  };
  size_t o_idx  = alloc((size_t)PTOT_ * 4);
  size_t o_xyzw = alloc((size_t)BN_ * 16);
  size_t o_W1h  = alloc(64 * 160 * 2);
  size_t o_W2h  = alloc(64 * 64 * 2);
  size_t o_Wvh  = alloc(16 * 160 * 2);
  size_t o_Wkh  = alloc(32 * 160 * 2);
  size_t o_Wqh  = alloc(256 * 64 * 2);
  size_t o_z1   = alloc((size_t)PTOT_ * 64 * 4);
  size_t o_z2   = alloc((size_t)PTOT_ * 64 * 4);
  size_t o_zk   = alloc((size_t)PTOT_ * 32 * 4);
  size_t o_zv   = alloc((size_t)PTOT_ * 16 * 4);
  size_t o_fqh  = alloc((size_t)BN_ * 64 * 2);
  size_t o_zq   = alloc((size_t)BN_ * 256 * 4);
  size_t o_p1s = alloc(SGRID * 64 * 4),  o_p1q = alloc(SGRID * 64 * 4);
  size_t o_p2s = alloc(SGRID * 64 * 4),  o_p2q = alloc(SGRID * 64 * 4);
  size_t o_pqs = alloc(SGRID * 256 * 4), o_pqq = alloc(SGRID * 256 * 4);
  size_t o_pvs = alloc(SGRID * 16 * 4),  o_pvq = alloc(SGRID * 16 * 4);
  size_t o_m1 = alloc(64 * 4),  o_r1 = alloc(64 * 4);
  size_t o_m2 = alloc(64 * 4),  o_r2 = alloc(64 * 4);
  size_t o_mq = alloc(256 * 4), o_rq = alloc(256 * 4);
  size_t o_mv = alloc(16 * 4),  o_rv = alloc(16 * 4);

  int*       idxp = (int*)(ws + o_idx);
  float4*    xyzw = (float4*)(ws + o_xyzw);
  _Float16*  W1h  = (_Float16*)(ws + o_W1h);
  _Float16*  W2h  = (_Float16*)(ws + o_W2h);
  _Float16*  Wvh  = (_Float16*)(ws + o_Wvh);
  _Float16*  Wkh  = (_Float16*)(ws + o_Wkh);
  _Float16*  Wqh  = (_Float16*)(ws + o_Wqh);
  float*     z1   = (float*)(ws + o_z1);
  float*     z2   = (float*)(ws + o_z2);
  float*     zk   = (float*)(ws + o_zk);
  float*     zv   = (float*)(ws + o_zv);
  _Float16*  fqh  = (_Float16*)(ws + o_fqh);
  float*     zq   = (float*)(ws + o_zq);
  float *p1s = (float*)(ws + o_p1s), *p1q = (float*)(ws + o_p1q);
  float *p2s = (float*)(ws + o_p2s), *p2q = (float*)(ws + o_p2q);
  float *pqs = (float*)(ws + o_pqs), *pqq = (float*)(ws + o_pqq);
  float *pvs = (float*)(ws + o_pvs), *pvq = (float*)(ws + o_pvq);
  float *m1 = (float*)(ws + o_m1), *r1 = (float*)(ws + o_r1);
  float *m2 = (float*)(ws + o_m2), *r2 = (float*)(ws + o_r2);
  float *mq = (float*)(ws + o_mq), *rq = (float*)(ws + o_rq);
  float *mv = (float*)(ws + o_mv), *rv = (float*)(ws + o_rv);

  float* out_xyz  = (float*)d_out;              // B*N*3
  float* out_attn = (float*)d_out + BN_ * 3;    // B*128*N

  const float invP  = 1.f / (float)PTOT_;
  const float invBN = 1.f / (float)BN_;

  k_convw<<<64, 256, 0, stream>>>(W1, W2, Wv, Wk, Wq, W1h, W2h, Wvh, Wkh, Wqh);
  k_pre<<<BN_ / 256, 256, 0, stream>>>(x, xyzw, out_xyz);
  k_knn<<<B_ * 32, 128, 0, stream>>>(xyzw, idxp);
  k_fcat_gemm<<<(int)(PTOT_ / 16), 128, 0, stream>>>(x, idxp, W1h, Wkh, Wvh, Wh, z1, zk, zv);
  k_stats<64><<<SGRID, 256, 0, stream>>>(z1, PTOT_ * 64, p1s, p1q);
  k_stats<16><<<SGRID, 256, 0, stream>>>(zv, PTOT_ * 16, pvs, pvq);
  k_finalize<<<1, 64, 0, stream>>>(p1s, p1q, 64, SGRID, invP, m1, r1);
  k_finalize<<<1, 16, 0, stream>>>(pvs, pvq, 16, SGRID, invP, mv, rv);
  k_bn1_gemm2<<<(int)(PTOT_ / 16), 128, 0, stream>>>(z1, m1, r1, g1, b1, W2h, z2);
  k_stats<64><<<SGRID, 256, 0, stream>>>(z2, PTOT_ * 64, p2s, p2q);
  k_finalize<<<1, 64, 0, stream>>>(p2s, p2q, 64, SGRID, invP, m2, r2);
  k_fqmax<<<BN_, 64, 0, stream>>>(z2, m2, r2, g2, b2, fqh);
  k_zq<<<BN_ / 16, 128, 0, stream>>>(fqh, Wqh, zq);
  k_stats<256><<<SGRID, 256, 0, stream>>>(zq, (long long)BN_ * 256, pqs, pqq);
  k_finalize<<<1, 256, 0, stream>>>(pqs, pqq, 256, SGRID, invBN, mq, rq);
  k_final<<<BN_, 128, 0, stream>>>(zq, mq, rq, gq, bq, zk, zv, mv, rv, gv, bv, out_attn);
}
